// MambaTTSDecoder_91311004713133
// MI455X (gfx1250) — compile-verified, matrix operand order checked by reading, not run on hardware
//
#include <hip/hip_runtime.h>
#include <hip/hip_bf16.h>

// ---------------- model constants ----------------
#define Dm    512
#define Lm    6
#define Hh    8
#define HD    64
#define DFFm  2048
#define Vv    1024
#define DIm   1024
#define DSm   16
#define DCONVm 4
#define DTRm  32
#define Bb    2
#define Tt    512
#define Ss    512
#define EPSf  1e-5f

// activation codes
#define ACT_NONE     0
#define ACT_GELU     1
#define ACT_SILU     2
#define ACT_SOFTPLUS 3

typedef __attribute__((ext_vector_type(16))) __bf16 bf16x16;
typedef __attribute__((ext_vector_type(8)))  float  f32x8;
typedef unsigned short ushort_t;

union FragU { struct { uint4 lo, hi; } q; bf16x16 v; };

__device__ __forceinline__ unsigned short f32_to_bf16(float f) {
    unsigned u = __float_as_uint(f);
    u += 0x7FFFu + ((u >> 16) & 1u);      // round-to-nearest-even
    return (unsigned short)(u >> 16);
}

__device__ __forceinline__ float silu_f(float x) { return x / (1.0f + __expf(-x)); }
__device__ __forceinline__ float gelu_f(float x) { return 0.5f * x * (1.0f + erff(x * 0.70710678118654752f)); }
__device__ __forceinline__ float softplus_f(float x) { return (x > 20.0f) ? x : log1pf(expf(x)); }

// ================= f32 -> bf16 bulk convert (one float4 per thread) =================
__global__ __launch_bounds__(256)
void cvt_bf16_kernel(const float* __restrict__ src, ushort_t* __restrict__ dst)
{
    int idx = blockIdx.x * 256 + threadIdx.x;
    float4 f = ((const float4*)src)[idx];
    uint2 p;
    p.x = (unsigned)f32_to_bf16(f.x) | ((unsigned)f32_to_bf16(f.y) << 16);
    p.y = (unsigned)f32_to_bf16(f.z) | ((unsigned)f32_to_bf16(f.w) << 16);
    *(uint2*)(dst + (size_t)idx * 4) = p;
}

// ================= generic WMMA GEMM (bf16 operands, f32 accumulate) =================
// C[M,N] = act( A[M,K](lda, bf16) @ W[N,K]^T(bf16) + bias[N] ) (+ res[M,N])
// Optional f32 (C) and/or bf16 (Cb) stores.
// BM=128, BK=32, BN = 32*NT (NT=4 -> 128 wide, NT=2 -> 64 wide).
// 256 threads = 8 waves; each wave -> 32 x (16*NT) output (2 x NT WMMA tiles).
// LDS rows padded to 40 bf16 (80B = 20 banks) -> conflict-free b128 fragment loads.
#define BM 128
#define BK 32
#define LDP 40

template<int NT>
__global__ __launch_bounds__(256)
void gemm_wmma_kernel(const ushort_t* __restrict__ A, int lda,
                      const ushort_t* __restrict__ W,
                      const float* __restrict__ bias,
                      const float* res,
                      float* C, ushort_t* Cb,
                      int M, int N, int K, int act)
{
    constexpr int BN = 32 * NT;
    __shared__ ushort_t As[BM * LDP];
    __shared__ ushort_t Ws[BN * LDP];

    const int tid  = threadIdx.x;
    const int lane = tid & 31;
    const int wave = tid >> 5;      // 0..7
    const int wm   = wave >> 1;     // 0..3 (M direction, 32 rows each)
    const int wn   = wave & 1;      // 0..1 (N direction, 16*NT cols each)
    const int bm0  = blockIdx.y * BM;
    const int bn0  = blockIdx.x * BN;

    const int rl    = lane & 15;    // row/col within 16
    const int khalf = lane >> 4;    // 0/1 -> K sub-halves per CDNA5 16-bit layout

    f32x8 acc[2][NT];
    #pragma unroll
    for (int i = 0; i < 2; ++i)
        #pragma unroll
        for (int j = 0; j < NT; ++j)
            #pragma unroll
            for (int e = 0; e < 8; ++e) acc[i][j][e] = 0.0f;

    for (int k0 = 0; k0 < K; k0 += BK) {
        __syncthreads();
        // ---- stage A tile: 128x32 bf16, 512 x b128 copies (2 per thread) ----
        #pragma unroll
        for (int i = 0; i < 2; ++i) {
            int e = tid + i * 256;            // 0..511
            int r = e >> 2;                   // 0..127
            int c = (e & 3) << 3;             // 0,8,16,24
            *(uint4*)(&As[r * LDP + c]) =
                *(const uint4*)(A + (size_t)(bm0 + r) * lda + k0 + c);
        }
        // ---- stage W tile: BNx32 bf16, BN*4 x b128 copies ----
        #pragma unroll
        for (int i = 0; i < BN / 64; ++i) {
            int e = tid + i * 256;            // 0..BN*4-1
            int r = e >> 2;                   // 0..BN-1
            int c = (e & 3) << 3;
            *(uint4*)(&Ws[r * LDP + c]) =
                *(const uint4*)(W + (size_t)(bn0 + r) * K + k0 + c);
        }
        __syncthreads();

        // ---- gather fragments per CDNA5 16-bit A/B lane layout ----
        FragU afr[2], bfr[NT];
        #pragma unroll
        for (int mt = 0; mt < 2; ++mt) {
            int arow = wm * 32 + mt * 16 + rl;
            afr[mt].q.lo = *(const uint4*)(&As[arow * LDP + khalf * 8]);
            afr[mt].q.hi = *(const uint4*)(&As[arow * LDP + 16 + khalf * 8]);
        }
        #pragma unroll
        for (int nt = 0; nt < NT; ++nt) {
            int brow = wn * (16 * NT) + nt * 16 + rl;   // W row == output column
            bfr[nt].q.lo = *(const uint4*)(&Ws[brow * LDP + khalf * 8]);
            bfr[nt].q.hi = *(const uint4*)(&Ws[brow * LDP + 16 + khalf * 8]);
        }
        #pragma unroll
        for (int mt = 0; mt < 2; ++mt)
            #pragma unroll
            for (int nt = 0; nt < NT; ++nt)
                acc[mt][nt] = __builtin_amdgcn_wmma_f32_16x16x32_bf16(
                    false, afr[mt].v, false, bfr[nt].v,
                    (short)0, acc[mt][nt], false, false);
    }

    // ---- epilogue (C layout: lane&15 = N, 8*(lane>>4)+j = M) ----
    #pragma unroll
    for (int mt = 0; mt < 2; ++mt) {
        #pragma unroll
        for (int nt = 0; nt < NT; ++nt) {
            int gn  = bn0 + wn * (16 * NT) + nt * 16 + rl;
            int gmb = bm0 + wm * 32 + mt * 16 + khalf * 8;
            float bv = bias ? bias[gn] : 0.0f;
            #pragma unroll
            for (int j = 0; j < 8; ++j) {
                int gm = gmb + j;
                float v = acc[mt][nt][j] + bv;
                if      (act == ACT_GELU)     v = gelu_f(v);
                else if (act == ACT_SILU)     v = silu_f(v);
                else if (act == ACT_SOFTPLUS) v = softplus_f(v);
                if (res) v += res[(size_t)gm * N + gn];
                if (C)  C[(size_t)gm * N + gn]  = v;
                if (Cb) Cb[(size_t)gm * N + gn] = f32_to_bf16(v);
            }
        }
    }
}

// ================= embedding =================
__global__ __launch_bounds__(256)
void embed_kernel(const float* __restrict__ tok_emb, const float* __restrict__ pos_emb,
                  const float* __restrict__ quant_emb, const int* __restrict__ tokens,
                  float* __restrict__ x)
{
    int row = blockIdx.x;               // b*T + t
    int t = row % Tt;
    int tok = tokens[row];
    for (int d = threadIdx.x; d < Dm; d += 256)
        x[(size_t)row * Dm + d] = tok_emb[(size_t)tok * Dm + d] + pos_emb[(size_t)t * Dm + d] + quant_emb[d];
}

// ================= layernorm (block per row) -> bf16 output =================
__global__ __launch_bounds__(256)
void ln_kernel(const float* __restrict__ x, const float* __restrict__ g,
               const float* __restrict__ b, ushort_t* __restrict__ o)
{
    __shared__ float r1[256], r2[256];
    int row = blockIdx.x;
    const float* xr = x + (size_t)row * Dm;
    float s = 0.f, s2 = 0.f;
    for (int i = threadIdx.x; i < Dm; i += 256) { float v = xr[i]; s += v; s2 += v * v; }
    r1[threadIdx.x] = s; r2[threadIdx.x] = s2;
    __syncthreads();
    for (int st = 128; st > 0; st >>= 1) {
        if (threadIdx.x < st) { r1[threadIdx.x] += r1[threadIdx.x + st]; r2[threadIdx.x] += r2[threadIdx.x + st]; }
        __syncthreads();
    }
    float mu  = r1[0] * (1.0f / Dm);
    float var = r2[0] * (1.0f / Dm) - mu * mu;
    float rs  = rsqrtf(var + EPSf);
    for (int i = threadIdx.x; i < Dm; i += 256)
        o[(size_t)row * Dm + i] = f32_to_bf16((xr[i] - mu) * rs * g[i] + b[i]);
}

// ================= depthwise causal conv + SiLU (f32 + bf16 outputs) =================
__global__ __launch_bounds__(256)
void conv_kernel(const float* __restrict__ xz, const float* __restrict__ cw,
                 const float* __restrict__ cb, float* __restrict__ xc,
                 ushort_t* __restrict__ xc16)
{
    int idx = blockIdx.x * 256 + threadIdx.x;      // over B*T*DI
    int c = idx % DIm;
    int t = (idx / DIm) % Tt;
    int b = idx / (DIm * Tt);
    float a = cb[c];
    #pragma unroll
    for (int k = 0; k < DCONVm; ++k) {
        int tt = t + k - (DCONVm - 1);
        if (tt >= 0)
            a += xz[((size_t)(b * Tt + tt)) * (2 * DIm) + c] * cw[c * DCONVm + k];
    }
    float v = silu_f(a);
    size_t o = (size_t)(b * Tt + t) * DIm + c;
    xc[o] = v;
    xc16[o] = f32_to_bf16(v);
}

// ================= selective scan (thread per (b, channel)) -> bf16 output ========
__global__ __launch_bounds__(256)
void scan_kernel(const float* __restrict__ xc, const float* __restrict__ dt,
                 const float* __restrict__ proj, const float* __restrict__ xz,
                 const float* __restrict__ A_log, const float* __restrict__ Dp,
                 ushort_t* __restrict__ ym)
{
    int id = blockIdx.x * 256 + threadIdx.x;   // 0 .. B*DI-1
    int b = id >> 10;
    int c = id & (DIm - 1);
    float Aneg[DSm], hst[DSm];
    #pragma unroll
    for (int s = 0; s < DSm; ++s) { Aneg[s] = -__expf(A_log[c * DSm + s]); hst[s] = 0.f; }
    float Dv = Dp[c];
    for (int t = 0; t < Tt; ++t) {
        size_t row = (size_t)(b * Tt + t);
        float dtv = dt[row * DIm + c];
        float xcv = xc[row * DIm + c];
        float u = dtv * xcv;
        const float* pr = proj + row * (DTRm + 2 * DSm);
        float y = 0.f;
        #pragma unroll
        for (int s = 0; s < DSm; ++s) {
            float dA = __expf(dtv * Aneg[s]);
            float h  = dA * hst[s] + u * pr[DTRm + s];          // Bc
            hst[s] = h;
            y += h * pr[DTRm + DSm + s];                        // Cc
        }
        float z = xz[row * (2 * DIm) + DIm + c];
        ym[row * DIm + c] = f32_to_bf16((y + Dv * xcv) * silu_f(z));
    }
}

// ================= cross-attention: block per (b,h,t) -> bf16 output =================
__global__ __launch_bounds__(128)
void attn_kernel(const float* __restrict__ q, const float* __restrict__ k,
                 const float* __restrict__ v, const unsigned char* __restrict__ mask,
                 ushort_t* __restrict__ o)
{
    __shared__ float qs[HD];
    __shared__ float ss[Ss];
    __shared__ float red[128];
    __shared__ float sinv;
    int t = blockIdx.x, h = blockIdx.y, b = blockIdx.z;
    int tid = threadIdx.x;
    if (tid < HD) qs[tid] = q[((size_t)(b * Tt + t)) * Dm + h * HD + tid];
    __syncthreads();
    // scores
    for (int j = tid; j < Ss; j += 128) {
        const float* kr = k + ((size_t)(b * Ss + j)) * Dm + h * HD;
        float d = 0.f;
        #pragma unroll
        for (int e = 0; e < HD; ++e) d += qs[e] * kr[e];
        float sv = d * 0.125f;                         // 1/sqrt(64)
        if (!mask[b * Ss + j]) sv = -1e9f;
        ss[j] = sv;
    }
    __syncthreads();
    // max reduce
    float lm = -1e30f;
    for (int j = tid; j < Ss; j += 128) lm = fmaxf(lm, ss[j]);
    red[tid] = lm; __syncthreads();
    for (int st = 64; st > 0; st >>= 1) {
        if (tid < st) red[tid] = fmaxf(red[tid], red[tid + st]);
        __syncthreads();
    }
    float mx = red[0];
    __syncthreads();
    // exp + sum reduce
    float lsum = 0.f;
    for (int j = tid; j < Ss; j += 128) { float e = __expf(ss[j] - mx); ss[j] = e; lsum += e; }
    red[tid] = lsum; __syncthreads();
    for (int st = 64; st > 0; st >>= 1) {
        if (tid < st) red[tid] += red[tid + st];
        __syncthreads();
    }
    if (tid == 0) sinv = 1.0f / red[0];
    __syncthreads();
    // weighted sum over V
    if (tid < HD) {
        float acc = 0.f;
        const float* vb = v + (size_t)b * Ss * Dm + h * HD + tid;
        for (int j = 0; j < Ss; ++j) acc += ss[j] * vb[(size_t)j * Dm];
        o[((size_t)(b * Tt + t)) * Dm + h * HD + tid] = f32_to_bf16(acc * sinv);
    }
}

// ================= host-side orchestration =================
static inline void gemm(hipStream_t st, const ushort_t* A, int lda, const ushort_t* W,
                        const float* bias, const float* res, float* C, ushort_t* Cb,
                        int M, int N, int K, int act)
{
    if (N % 128 == 0)
        gemm_wmma_kernel<4><<<dim3(N / 128, M / BM), 256, 0, st>>>(A, lda, W, bias, res, C, Cb, M, N, K, act);
    else
        gemm_wmma_kernel<2><<<dim3(N / 64,  M / BM), 256, 0, st>>>(A, lda, W, bias, res, C, Cb, M, N, K, act);
}

static inline void cvt(hipStream_t st, const float* src, ushort_t* dst, int n)
{
    cvt_bf16_kernel<<<n / 1024, 256, 0, st>>>(src, dst);   // 4 elements per thread
}

extern "C" void kernel_launch(void* const* d_in, const int* in_sizes, int n_in,
                              void* d_out, int out_size, void* d_ws, size_t ws_size,
                              hipStream_t stream)
{
    const float* styled    = (const float*)d_in[0];
    const float* tok_emb   = (const float*)d_in[1];
    const float* pos_emb   = (const float*)d_in[2];
    const float* quant_emb = (const float*)d_in[3];
    const float* ln_m_g    = (const float*)d_in[4];
    const float* ln_m_b    = (const float*)d_in[5];
    const float* in_proj_w = (const float*)d_in[6];
    const float* conv_w    = (const float*)d_in[7];
    const float* conv_b    = (const float*)d_in[8];
    const float* xproj_w   = (const float*)d_in[9];
    const float* dtproj_w  = (const float*)d_in[10];
    const float* dtproj_b  = (const float*)d_in[11];
    const float* A_log     = (const float*)d_in[12];
    const float* D_param   = (const float*)d_in[13];
    const float* outproj_w = (const float*)d_in[14];
    const float* ln_c_g    = (const float*)d_in[15];
    const float* ln_c_b    = (const float*)d_in[16];
    const float* wq        = (const float*)d_in[17];
    const float* bq        = (const float*)d_in[18];
    const float* wk        = (const float*)d_in[19];
    const float* bk        = (const float*)d_in[20];
    const float* wv        = (const float*)d_in[21];
    const float* bv        = (const float*)d_in[22];
    const float* wo        = (const float*)d_in[23];
    const float* bo        = (const float*)d_in[24];
    const float* ln_f_g    = (const float*)d_in[25];
    const float* ln_f_b    = (const float*)d_in[26];
    const float* ff_w1     = (const float*)d_in[27];
    const float* ff_b1     = (const float*)d_in[28];
    const float* ff_w2     = (const float*)d_in[29];
    const float* ff_b2     = (const float*)d_in[30];
    const float* lnout_g   = (const float*)d_in[31];
    const float* lnout_b   = (const float*)d_in[32];
    const float* head_w    = (const float*)d_in[33];
    const float* head_b    = (const float*)d_in[34];
    const int*   tokens    = (const int*)d_in[35];
    const unsigned char* smask = (const unsigned char*)d_in[36];

    const int M = Bb * Tt;  // 1024 rows for all token-major GEMMs (== B*S too)

    // ---- f32 scratch ----
    float* ws   = (float*)d_ws;
    float* x    = ws;                  // [M, D]       524288
    float* big  = ws + 524288;         // [M, 2DI] xz  2097152
    float* xcb  = ws + 2621440;        // [M, DI]      1048576
    float* prj  = ws + 3670016;        // [M, 64]      65536
    float* dtb  = ws + 3735552;        // [M, DI]      1048576
    float* qb   = ws + 4784128;        // [M, D]       524288
    float* kb   = ws + 5308416;        // [M, D]       524288
    float* vb2  = ws + 5832704;        // [M, D]       524288 -> ends 6356992 floats
    // ---- bf16 scratch ----
    ushort_t* bws      = (ushort_t*)(ws + 6356992);
    ushort_t* hb16     = bws;                 // [M, D]      524288
    ushort_t* xc16     = bws + 524288;        // [M, DI]     1048576
    ushort_t* prj16    = bws + 1572864;       // [M, 64]     65536
    ushort_t* ym16     = bws + 1638400;       // [M, DI]     1048576
    ushort_t* ao16     = bws + 2686976;       // [M, D]      524288
    ushort_t* ff16     = bws + 3211264;       // [M, DFF]    2097152
    ushort_t* styled16 = bws + 5308416;       // [M, D]      524288
    ushort_t* warena   = bws + 5832704;       // max weight  1048576 -> ends 6881280 (~39 MB total)

    embed_kernel<<<M, 256, 0, stream>>>(tok_emb, pos_emb, quant_emb, tokens, x);
    cvt(stream, styled, styled16, M * Dm);

    for (int l = 0; l < Lm; ++l) {
        // ---------- Mamba block ----------
        ln_kernel<<<M, 256, 0, stream>>>(x, ln_m_g + l * Dm, ln_m_b + l * Dm, hb16);
        cvt(stream, in_proj_w + (size_t)l * 2 * DIm * Dm, warena, 2 * DIm * Dm);
        gemm(stream, hb16, Dm, warena, nullptr, nullptr, big, nullptr,
             M, 2 * DIm, Dm, ACT_NONE);
        conv_kernel<<<(M * DIm) / 256, 256, 0, stream>>>(big, conv_w + (size_t)l * DIm * DCONVm,
                                                         conv_b + (size_t)l * DIm, xcb, xc16);
        cvt(stream, xproj_w + (size_t)l * (DTRm + 2 * DSm) * DIm, warena, (DTRm + 2 * DSm) * DIm);
        gemm(stream, xc16, DIm, warena, nullptr, nullptr, prj, prj16,
             M, DTRm + 2 * DSm, DIm, ACT_NONE);
        cvt(stream, dtproj_w + (size_t)l * DIm * DTRm, warena, DIm * DTRm);
        gemm(stream, prj16, DTRm + 2 * DSm, warena, dtproj_b + (size_t)l * DIm, nullptr,
             dtb, nullptr, M, DIm, DTRm, ACT_SOFTPLUS);
        scan_kernel<<<(Bb * DIm) / 256, 256, 0, stream>>>(xcb, dtb, prj, big,
                                                          A_log + (size_t)l * DIm * DSm,
                                                          D_param + (size_t)l * DIm, ym16);
        cvt(stream, outproj_w + (size_t)l * Dm * DIm, warena, Dm * DIm);
        gemm(stream, ym16, DIm, warena, nullptr, x, x, nullptr, M, Dm, DIm, ACT_NONE);

        // ---------- cross-attention ----------
        ln_kernel<<<M, 256, 0, stream>>>(x, ln_c_g + l * Dm, ln_c_b + l * Dm, hb16);
        cvt(stream, wq + (size_t)l * Dm * Dm, warena, Dm * Dm);
        gemm(stream, hb16, Dm, warena, bq + l * Dm, nullptr, qb, nullptr, M, Dm, Dm, ACT_NONE);
        cvt(stream, wk + (size_t)l * Dm * Dm, warena, Dm * Dm);
        gemm(stream, styled16, Dm, warena, bk + l * Dm, nullptr, kb, nullptr, M, Dm, Dm, ACT_NONE);
        cvt(stream, wv + (size_t)l * Dm * Dm, warena, Dm * Dm);
        gemm(stream, styled16, Dm, warena, bv + l * Dm, nullptr, vb2, nullptr, M, Dm, Dm, ACT_NONE);
        attn_kernel<<<dim3(Tt, Hh, Bb), 128, 0, stream>>>(qb, kb, vb2, smask, ao16);
        cvt(stream, wo + (size_t)l * Dm * Dm, warena, Dm * Dm);
        gemm(stream, ao16, Dm, warena, bo + l * Dm, x, x, nullptr, M, Dm, Dm, ACT_NONE);

        // ---------- FFN ----------
        ln_kernel<<<M, 256, 0, stream>>>(x, ln_f_g + l * Dm, ln_f_b + l * Dm, hb16);
        cvt(stream, ff_w1 + (size_t)l * DFFm * Dm, warena, DFFm * Dm);
        gemm(stream, hb16, Dm, warena, ff_b1 + (size_t)l * DFFm, nullptr,
             nullptr, ff16, M, DFFm, Dm, ACT_GELU);
        cvt(stream, ff_w2 + (size_t)l * Dm * DFFm, warena, Dm * DFFm);
        gemm(stream, ff16, DFFm, warena, ff_b2 + l * Dm, x, x, nullptr, M, Dm, DFFm, ACT_NONE);
    }

    ln_kernel<<<M, 256, 0, stream>>>(x, lnout_g, lnout_b, hb16);
    cvt(stream, head_w, warena, Vv * Dm);
    gemm(stream, hb16, Dm, warena, head_b, nullptr, (float*)d_out, nullptr, M, Vv, Dm, ACT_NONE);
}
